// MoE_Layer_2207613190858
// MI455X (gfx1250) — matrix operand order repaired and checked by reading, not compiled, not run on hardware
//
#include <hip/hip_runtime.h>
#include <stdint.h>

// ---------------- problem constants ----------------
#define B_TOK 8192
#define D_DIM 1280
#define E_EXP 4
#define NCHK  20          // NC
#define CDIM  64          // CD
#define FFD   256         // FF
#define QKV   192         // 3*CD

// ---------------- WMMA types -----------------------
typedef __bf16  bf16_t;
typedef bf16_t  v16bf __attribute__((ext_vector_type(16)));
typedef bf16_t  v2bf  __attribute__((ext_vector_type(2)));
typedef float   v8f   __attribute__((ext_vector_type(8)));
typedef unsigned int u32x4 __attribute__((ext_vector_type(4)));
typedef int     i32x8 __attribute__((ext_vector_type(8)));
typedef int     i32x4 __attribute__((ext_vector_type(4)));

union Frag { unsigned int u[8]; v16bf v; };

__device__ __forceinline__ unsigned short f2bf(float f) {
  unsigned int u = __float_as_uint(f);
  u += 0x7FFFu + ((u >> 16) & 1u);              // round-to-nearest-even
  return (unsigned short)(u >> 16);
}
// packed f32x2 -> bf16x2 (hardware v_cvt_pk_bf16_f32 when available)
__device__ __forceinline__ unsigned int packbf(float lo, float hi) {
#if __has_builtin(__builtin_amdgcn_cvt_pk_bf16_f32)
  union { v2bf v; unsigned int u; } c;
  c.v = __builtin_amdgcn_cvt_pk_bf16_f32(lo, hi);
  return c.u;
#else
  return (unsigned int)f2bf(lo) | ((unsigned int)f2bf(hi) << 16);
#endif
}
// A-fragment (16x32 bf16) K index for VGPR v, lane l (contiguous: k = 16*(lane/16) + 2v)
__device__ __forceinline__ int a_kidx(int v, int lane) {
  return (((lane >> 4) & 1) << 4) + (v << 1);
}

// ---------------- zero-init output ------------------
__global__ void zero_kernel(float* __restrict__ p, int n) {
  int i = blockIdx.x * blockDim.x + threadIdx.x;
  if (i < n) p[i] = 0.f;
}

// ---------------- trunk weights f32 -> bf16 ---------
// layout per expert (elements): ipw[12288] | opw[4096] | fw1[16384] | fw2[16384]
__global__ void convert_weights_kernel(const float* __restrict__ ipw,
                                       const float* __restrict__ opw,
                                       const float* __restrict__ fw1,
                                       const float* __restrict__ fw2,
                                       unsigned short* __restrict__ wbf) {
  int i = blockIdx.x * blockDim.x + threadIdx.x;
  const int per = 49152;
  if (i >= E_EXP * per) return;
  int e = i / per, r = i % per;
  float v;
  if (r < 12288)       v = ipw[e * 12288 + r];
  else if (r < 16384)  v = opw[e * 4096  + (r - 12288)];
  else if (r < 32768)  v = fw1[e * 16384 + (r - 16384)];
  else                 v = fw2[e * 16384 + (r - 32768)];
  wbf[i] = f2bf(v);
}

// ---------------- proj_w f32 -> bf16 ----------------
__global__ void convert_pw_kernel(const float* __restrict__ pw,
                                  unsigned short* __restrict__ pwbf) {
  size_t i = (size_t)blockIdx.x * blockDim.x + threadIdx.x;
  if (i < (size_t)E_EXP * D_DIM * D_DIM) pwbf[i] = f2bf(pw[i]);
}

// ---------------- gating: one wave per token --------
__global__ void gate_kernel(const float* __restrict__ x,
                            const float* __restrict__ gw1, const float* __restrict__ gb1,
                            const float* __restrict__ gw2, const float* __restrict__ gb2,
                            float* __restrict__ wout, float* __restrict__ outTail) {
  const int lane = threadIdx.x & 31;
  const int b = blockIdx.x * (blockDim.x >> 5) + (threadIdx.x >> 5);
  if (b >= B_TOK) return;
  float acc[16];
#pragma unroll
  for (int j = 0; j < 16; ++j) acc[j] = 0.f;
  const float* xr = x + (size_t)b * D_DIM;
  for (int d = lane; d < D_DIM; d += 32) {
    float xv = xr[d];
#pragma unroll
    for (int j = 0; j < 16; ++j) acc[j] += xv * gw1[j * D_DIM + d];
  }
#pragma unroll
  for (int j = 0; j < 16; ++j) {
    for (int off = 16; off >= 1; off >>= 1) acc[j] += __shfl_xor(acc[j], off, 32);
    acc[j] = tanhf(acc[j] + gb1[j]);
  }
  float lg[4];
#pragma unroll
  for (int e = 0; e < 4; ++e) {
    float s = gb2[e];
#pragma unroll
    for (int j = 0; j < 16; ++j) s += acc[j] * gw2[e * 16 + j];
    lg[e] = s;
  }
  float mx = fmaxf(fmaxf(lg[0], lg[1]), fmaxf(lg[2], lg[3]));
  float sm[4], se = 0.f;
#pragma unroll
  for (int e = 0; e < 4; ++e) { sm[e] = __expf(lg[e] - mx); se += sm[e]; }
#pragma unroll
  for (int e = 0; e < 4; ++e) sm[e] /= se;
  int i0 = 0;
#pragma unroll
  for (int e = 1; e < 4; ++e) if (sm[e] > sm[i0]) i0 = e;
  int i1 = (i0 == 0) ? 1 : 0;
#pragma unroll
  for (int e = 0; e < 4; ++e) if (e != i1 && e != i0 && sm[e] > sm[i1]) i1 = e;
  float denom = sm[i0] + sm[i1] + 1e-6f;
  if (lane == 0) {
    float w4[4] = {0.f, 0.f, 0.f, 0.f};
    w4[i0] = sm[i0] / denom;
    w4[i1] = sm[i1] / denom;
#pragma unroll
    for (int e = 0; e < 4; ++e) wout[b * E_EXP + e] = w4[e];
    atomicAdd(outTail + 1 + i0, 1.f);   // load[E]; outTail[0] = aux_loss = 0
    atomicAdd(outTail + 1 + i1, 1.f);
  }
}

// ---------------- per-wave trunk GEMM helpers -------
// M = 20 rows: M-tile0 (rows 0..15) is fully valid; M-tile1 only rows 16..19
// (A rows guarded by ml<4; epilogue guarded by a single hi==0 region, v<4).
// K=64 variant: A fragments (2 M-tiles x 2 K-steps) hoisted out of the N loop.
template <typename Epi>
__device__ __forceinline__ void wave_gemm64(const unsigned short* A, int as,
                                            const unsigned short* W,
                                            int Ntiles, int lane, Epi epi) {
  const int ml = lane & 15;
  const int hi = (lane >> 4) & 1;
  const unsigned int* A32 = (const unsigned int*)A;
  Frag af[2][2];                       // [mt][ks], 32 VGPRs
#pragma unroll
  for (int ks = 0; ks < 2; ++ks) {
#pragma unroll
    for (int v = 0; v < 8; ++v)
      af[0][ks].u[v] = A32[(ml * as + ks * 32 + a_kidx(v, lane)) >> 1];
    if (ml < 4) {
#pragma unroll
      for (int v = 0; v < 8; ++v)
        af[1][ks].u[v] = A32[((16 + ml) * as + ks * 32 + a_kidx(v, lane)) >> 1];
    } else {
#pragma unroll
      for (int v = 0; v < 8; ++v) af[1][ks].u[v] = 0u;   // M padding rows -> 0
    }
  }
  for (int nt = 0; nt < Ntiles; ++nt) {
    const int n = nt * 16 + ml;
    const unsigned int* wrow = (const unsigned int*)(W + n * 64);
    Frag b0, b1;
#pragma unroll
    for (int v = 0; v < 8; ++v) {
      b0.u[v] = wrow[(hi << 3) + v];
      b1.u[v] = wrow[16 + (hi << 3) + v];
    }
    v8f acc0 = {0.f, 0.f, 0.f, 0.f, 0.f, 0.f, 0.f, 0.f};
    v8f acc1 = {0.f, 0.f, 0.f, 0.f, 0.f, 0.f, 0.f, 0.f};
    acc0 = __builtin_amdgcn_wmma_f32_16x16x32_bf16(false, af[0][0].v, false, b0.v,
                                                   (short)0, acc0, false, false);
    acc0 = __builtin_amdgcn_wmma_f32_16x16x32_bf16(false, af[0][1].v, false, b1.v,
                                                   (short)0, acc0, false, false);
    acc1 = __builtin_amdgcn_wmma_f32_16x16x32_bf16(false, af[1][0].v, false, b0.v,
                                                   (short)0, acc1, false, false);
    acc1 = __builtin_amdgcn_wmma_f32_16x16x32_bf16(false, af[1][1].v, false, b1.v,
                                                   (short)0, acc1, false, false);
#pragma unroll
    for (int v = 0; v < 8; ++v) epi((hi << 3) + v, n, acc0[v]);  // rows 0..15
    if (hi == 0) {
#pragma unroll
      for (int v = 0; v < 4; ++v) epi(16 + v, n, acc1[v]);       // rows 16..19
    }
  }
}

// generic-K variant (used for ffn2, K=256); B fragment shared by both M-tiles
template <typename Epi>
__device__ __forceinline__ void wave_gemm(const unsigned short* A, int as, int K,
                                          const unsigned short* W,
                                          int Ntiles, int lane, Epi epi) {
  const int ml = lane & 15;
  const int hi = (lane >> 4) & 1;
  const unsigned int* A32 = (const unsigned int*)A;
  for (int nt = 0; nt < Ntiles; ++nt) {
    const int n = nt * 16 + ml;
    const unsigned int* wrow = (const unsigned int*)(W + n * K);
    v8f acc0 = {0.f, 0.f, 0.f, 0.f, 0.f, 0.f, 0.f, 0.f};
    v8f acc1 = {0.f, 0.f, 0.f, 0.f, 0.f, 0.f, 0.f, 0.f};
    for (int ks = 0; ks < (K >> 5); ++ks) {
      Frag a0, a1, bf;
#pragma unroll
      for (int v = 0; v < 8; ++v)
        a0.u[v] = A32[(ml * as + ks * 32 + a_kidx(v, lane)) >> 1];
      if (ml < 4) {
#pragma unroll
        for (int v = 0; v < 8; ++v)
          a1.u[v] = A32[((16 + ml) * as + ks * 32 + a_kidx(v, lane)) >> 1];
      } else {
#pragma unroll
        for (int v = 0; v < 8; ++v) a1.u[v] = 0u;
      }
#pragma unroll
      for (int v = 0; v < 8; ++v) bf.u[v] = wrow[(ks << 4) + (hi << 3) + v];
      acc0 = __builtin_amdgcn_wmma_f32_16x16x32_bf16(false, a0.v, false, bf.v,
                                                     (short)0, acc0, false, false);
      acc1 = __builtin_amdgcn_wmma_f32_16x16x32_bf16(false, a1.v, false, bf.v,
                                                     (short)0, acc1, false, false);
    }
#pragma unroll
    for (int v = 0; v < 8; ++v) epi((hi << 3) + v, n, acc0[v]);
    if (hi == 0) {
#pragma unroll
      for (int v = 0; v < 4; ++v) epi(16 + v, n, acc1[v]);
    }
  }
}

// ---------------- expert trunk: one wave per token --
// f32 state: xs (residual), sc (qkv, read by VALU attention)
// bf16 state (WMMA A operands): xnb (xn / o / xn2), hb (ffn hidden)
#define XS_S 66      // f32 & bf16 row stride (even)
#define SC_S 194     // qkv f32 row stride
#define HB_S 258     // ffn-hidden bf16 row stride (even)

__global__ void __launch_bounds__(32)
trunk_kernel(const float* __restrict__ x,   const float* __restrict__ pos,
             const float* __restrict__ g1,  const float* __restrict__ ipb,
             const float* __restrict__ opb, const float* __restrict__ ls1,
             const float* __restrict__ g2,  const float* __restrict__ fb1,
             const float* __restrict__ fb2, const float* __restrict__ ls2,
             const unsigned short* __restrict__ wbf,
             const float* __restrict__ wgate,
             unsigned short* __restrict__ act, int e) {
  __shared__ float xs[NCHK * XS_S];            //  5,280 B
  __shared__ float sc[NCHK * SC_S];            // 15,520 B
  __shared__ unsigned short xnb[NCHK * XS_S];  //  2,640 B
  __shared__ unsigned short hb[NCHK * HB_S];   // 10,320 B
  const int lane = threadIdx.x & 31;
  const int b = blockIdx.x;

  const float wbe = wgate[b * E_EXP + e];
  unsigned short* actr = act + (size_t)b * D_DIM;
  if (wbe == 0.f) {                            // expert not selected: contribution is 0
    for (int i = lane * 2; i < D_DIM; i += 64) *(unsigned int*)(actr + i) = 0u;
    return;
  }
  const unsigned short* ipw = wbf + e * 49152;
  const unsigned short* opw = ipw + 12288;
  const unsigned short* fw1 = ipw + 16384;
  const unsigned short* fw2 = ipw + 32768;
  const float* ipbp = ipb + e * QKV;
  const float* opbp = opb + e * CDIM;
  const float* ls1p = ls1 + e * CDIM;
  const float* ls2p = ls2 + e * CDIM;
  const float* fb1p = fb1 + e * FFD;
  const float* fb2p = fb2 + e * CDIM;
  const float* g1p  = g1  + e * CDIM;
  const float* g2p  = g2  + e * CDIM;

  // 1) xs = x.reshape(NC,CD) + pos
  {
    const float* xr = x + (size_t)b * D_DIM;
    const float* pr = pos + (size_t)e * D_DIM;
    for (int i = lane; i < D_DIM; i += 32)
      xs[(i >> 6) * XS_S + (i & 63)] = xr[i] + pr[i];
  }
  // 2) xnb = bf16(rms(xs) * g)  -- written once, consumed as WMMA A operand
  auto rms = [&](const float* gv) {
    if (lane < NCHK) {
      float ss = 0.f;
#pragma unroll
      for (int k = 0; k < CDIM; ++k) { float t = xs[lane * XS_S + k]; ss += t * t; }
      float inv = 1.f / (sqrtf(ss) * 0.125f + 1e-8f);
#pragma unroll
      for (int k = 0; k < CDIM; k += 2)
        *(unsigned int*)(xnb + lane * XS_S + k) =
            packbf(xs[lane * XS_S + k] * inv * gv[k],
                   xs[lane * XS_S + k + 1] * inv * gv[k + 1]);
    }
  };
  rms(g1p);
  // 3) qkv = xn @ ipw.T + ipb   -> sc[:, 0:192] (f32, for VALU attention)
  wave_gemm64(xnb, XS_S, ipw, 12, lane,
              [&](int m, int n, float v) { sc[m * SC_S + n] = v + ipbp[n]; });
  // 4) attention (VALU; 4 heads x 20x20x16), o -> xnb (bf16)
  for (int p = lane; p < NCHK * 4; p += 32) {
    const int h = p / NCHK, qi = p % NCHK;
    float qv[16];
#pragma unroll
    for (int d = 0; d < 16; ++d) qv[d] = sc[qi * SC_S + h * 16 + d];
    float s[NCHK], mx = -3.0e38f;
#pragma unroll
    for (int ki = 0; ki < NCHK; ++ki) {
      float dot = 0.f;
#pragma unroll
      for (int d = 0; d < 16; ++d) dot += qv[d] * sc[ki * SC_S + 64 + h * 16 + d];
      s[ki] = dot * 0.25f;
      mx = fmaxf(mx, s[ki]);
    }
    float sum = 0.f;
#pragma unroll
    for (int ki = 0; ki < NCHK; ++ki) { s[ki] = __expf(s[ki] - mx); sum += s[ki]; }
    float inv = 1.f / sum;
#pragma unroll
    for (int d = 0; d < 16; d += 2) {
      float o0 = 0.f, o1 = 0.f;
#pragma unroll
      for (int ki = 0; ki < NCHK; ++ki) {
        o0 += s[ki] * sc[ki * SC_S + 128 + h * 16 + d];
        o1 += s[ki] * sc[ki * SC_S + 128 + h * 16 + d + 1];
      }
      *(unsigned int*)(xnb + qi * XS_S + h * 16 + d) = packbf(o0 * inv, o1 * inv);
    }
  }
  // 5) xs += ls1 * (o @ opw.T + opb)
  wave_gemm64(xnb, XS_S, opw, 4, lane,
              [&](int m, int n, float v) { xs[m * XS_S + n] += ls1p[n] * (v + opbp[n]); });
  // 6) xnb = bf16(rms(xs) * g2)
  rms(g2p);
  // 7) hb = bf16(relu(xn2 @ fw1.T + fb1))
  wave_gemm64(xnb, XS_S, fw1, 16, lane,
              [&](int m, int n, float v) {
                float t = v + fb1p[n];
                hb[m * HB_S + n] = f2bf(t > 0.f ? t : 0.f);
              });
  // 8) xs += ls2 * (h @ fw2.T + fb2)
  wave_gemm(hb, HB_S, FFD, fw2, 4, lane,
            [&](int m, int n, float v) { xs[m * XS_S + n] += ls2p[n] * (v + fb2p[n]); });
  // 9) act[b,:] = bf16(w[b,e] * xs), packed pairs
  for (int i = lane * 2; i < D_DIM; i += 64) {
    const int c = i >> 6, k = i & 63;
    *(unsigned int*)(actr + i) =
        packbf(wbe * xs[c * XS_S + k], wbe * xs[c * XS_S + k + 1]);
  }
}

// ---------------- final proj GEMM (accumulating) ----
// out[M,N] += act_bf16[M,K] @ pwbf[N,K]^T + w[m,e]*pb[n]
// Double-buffered LDS; BOTH tiles fetched by the Tensor Data Mover.
#define GM 128
#define GN 128
#define GK 32
#define KS_N (D_DIM / GK)   // 40

__global__ void __launch_bounds__(256)
gemm_kernel(const unsigned short* __restrict__ act,
            const unsigned short* __restrict__ pwbf, const float* __restrict__ pb,
            const float* __restrict__ wgate, float* __restrict__ out, int e) {
  __shared__ unsigned short As[2][GM * GK];   // 2 x 8 KB (TDM destination)
  __shared__ unsigned short Ws[2][GN * GK];   // 2 x 8 KB (TDM destination)
  const int tid = threadIdx.x;
  const int lane = tid & 31;
  const int wid  = tid >> 5;
  const int bn = blockIdx.x;               // N tile: 0..9
  const int bm = blockIdx.y;               // M tile: 0..63
  const int wm = wid & 3;                  // 4x2 wave grid -> 32(M) x 64(N) per wave
  const int wn = wid >> 2;
  const int ml = lane & 15, hi = lane >> 4;

  v8f acc[2][4];
#pragma unroll
  for (int i = 0; i < 2; ++i)
#pragma unroll
    for (int j = 0; j < 4; ++j) acc[i][j] = (v8f){0.f,0.f,0.f,0.f,0.f,0.f,0.f,0.f};

  const unsigned long long gaA =
      (unsigned long long)(uintptr_t)(act + (size_t)bm * GM * D_DIM);
  const unsigned long long gaW =
      (unsigned long long)(uintptr_t)(pwbf + (size_t)e * D_DIM * D_DIM
                                           + (size_t)bn * GN * D_DIM);
  // TDM: 2D tile 128(rows) x 32(bf16), row stride D_DIM elements
  auto tdm_issue = [&](unsigned long long base, int ks, unsigned short* lds) {
    unsigned long long ga = base + (unsigned long long)(ks * GK) * 2ull;
    u32x4 g0;
    g0.x = 1u;                                                    // count=1, user desc
    g0.y = (unsigned int)(uintptr_t)(void*)lds;                   // LDS dest
    g0.z = (unsigned int)(ga & 0xFFFFFFFFull);                    // global_addr lo
    g0.w = (unsigned int)((ga >> 32) & 0x1FFFFFFull) | (2u << 30);// addr hi | type=2
    i32x8 g1;
    g1[0] = 0x10000;            // wg_mask=0, data_size=1 (2 bytes)
    g1[1] = (int)(32u << 16);   // tensor_dim0 = 32
    g1[2] = (int)(128u << 16);  // tensor_dim1 = 128
    g1[3] = (int)(32u << 16);   // tile_dim0 = 32
    g1[4] = 128;                // tile_dim1 = 128, tile_dim2 = 0 (2D)
    g1[5] = D_DIM;              // tensor_dim0_stride = 1280
    g1[6] = 0; g1[7] = 0;
    i32x4 gz4 = {0, 0, 0, 0};
    i32x8 gz8 = {0, 0, 0, 0, 0, 0, 0, 0};
    __builtin_amdgcn_tensor_load_to_lds(g0, g1, gz4, gz4, gz8, 0);
  };
  auto compute = [&](int buf) {
    const unsigned int* As32 = (const unsigned int*)&As[buf][0];
    const unsigned int* Ws32 = (const unsigned int*)&Ws[buf][0];
    Frag a0, a1;
    const int m0 = wm * 32 + ml;
#pragma unroll
    for (int v = 0; v < 8; ++v) {
      const int k = a_kidx(v, lane);
      a0.u[v] = As32[m0 * 16 + (k >> 1)];
      a1.u[v] = As32[(m0 + 16) * 16 + (k >> 1)];
    }
#pragma unroll
    for (int nf = 0; nf < 4; ++nf) {
      Frag bf;
      const int n = wn * 64 + nf * 16 + ml;
#pragma unroll
      for (int v = 0; v < 8; ++v) bf.u[v] = Ws32[n * 16 + hi * 8 + v];
      acc[0][nf] = __builtin_amdgcn_wmma_f32_16x16x32_bf16(false, a0.v, false, bf.v,
                                                           (short)0, acc[0][nf], false, false);
      acc[1][nf] = __builtin_amdgcn_wmma_f32_16x16x32_bf16(false, a1.v, false, bf.v,
                                                           (short)0, acc[1][nf], false, false);
    }
  };

  // software pipeline: TDM(k+1) overlaps compute(k); one barrier per step
  if (wid == 0) {
    tdm_issue(gaA, 0, &As[0][0]);
    tdm_issue(gaW, 0, &Ws[0][0]);
  }
  for (int ks = 0; ks < KS_N; ++ks) {
    const int cur = ks & 1;
    if (wid == 0) __builtin_amdgcn_s_wait_tensorcnt(0);  // tiles(ks) complete
    __syncthreads();                             // tiles[cur] ready; prev compute done
    if (ks + 1 < KS_N && wid == 0) {             // safe: [cur^1] readers finished
      tdm_issue(gaA, ks + 1, &As[cur ^ 1][0]);
      tdm_issue(gaW, ks + 1, &Ws[cur ^ 1][0]);
    }
    compute(cur);
  }
  // --- epilogue: out += acc + w[m,e]*pb[n] ---
  const float* pbe = pb + e * D_DIM;
#pragma unroll
  for (int mf = 0; mf < 2; ++mf)
#pragma unroll
    for (int nf = 0; nf < 4; ++nf)
#pragma unroll
      for (int v = 0; v < 8; ++v) {
        const int m = bm * GM + wm * 32 + mf * 16 + v + hi * 8;
        const int n = bn * GN + wn * 64 + nf * 16 + ml;
        const float bias = wgate[m * E_EXP + e] * pbe[n];
        out[(size_t)m * D_DIM + n] += acc[mf][nf][v] + bias;
      }
}

// ---------------- host-side launch ------------------
extern "C" void kernel_launch(void* const* d_in, const int* in_sizes, int n_in,
                              void* d_out, int out_size, void* d_ws, size_t ws_size,
                              hipStream_t stream) {
  const float* x    = (const float*)d_in[0];
  const float* gw1  = (const float*)d_in[1];
  const float* gb1  = (const float*)d_in[2];
  const float* gw2  = (const float*)d_in[3];
  const float* gb2  = (const float*)d_in[4];
  const float* pos  = (const float*)d_in[5];
  const float* n1g  = (const float*)d_in[6];
  const float* ipw  = (const float*)d_in[7];
  const float* ipb  = (const float*)d_in[8];
  const float* opw  = (const float*)d_in[9];
  const float* opb  = (const float*)d_in[10];
  const float* ls1  = (const float*)d_in[11];
  const float* n2g  = (const float*)d_in[12];
  const float* fw1  = (const float*)d_in[13];
  const float* fb1  = (const float*)d_in[14];
  const float* fw2  = (const float*)d_in[15];
  const float* fb2  = (const float*)d_in[16];
  const float* ls2  = (const float*)d_in[17];
  const float* pw   = (const float*)d_in[18];
  const float* pb   = (const float*)d_in[19];
  float* out = (float*)d_out;

  // workspace: w[B,E] f32 | act[B,D] bf16 | trunk weights bf16 | proj_w bf16
  float* wgate = (float*)d_ws;
  unsigned short* act  = (unsigned short*)((char*)d_ws + (size_t)B_TOK * E_EXP * 4);
  unsigned short* wbf  = (unsigned short*)((char*)act + (size_t)B_TOK * D_DIM * 2);
  unsigned short* pwbf = (unsigned short*)((char*)wbf + (size_t)E_EXP * 49152 * 2);

  zero_kernel<<<(out_size + 255) / 256, 256, 0, stream>>>(out, out_size);
  convert_weights_kernel<<<(E_EXP * 49152 + 255) / 256, 256, 0, stream>>>(ipw, opw, fw1, fw2, wbf);
  convert_pw_kernel<<<(E_EXP * D_DIM * D_DIM + 255) / 256, 256, 0, stream>>>(pw, pwbf);
  gate_kernel<<<B_TOK / 8, 256, 0, stream>>>(x, gw1, gb1, gw2, gb2, wgate,
                                             out + (size_t)B_TOK * D_DIM);
  for (int e = 0; e < E_EXP; ++e) {
    trunk_kernel<<<B_TOK, 32, 0, stream>>>(
        x, pos, n1g, ipb, opb, ls1, n2g, fb1, fb2, ls2, wbf, wgate, act, e);
    gemm_kernel<<<dim3(D_DIM / GN, B_TOK / GM), 256, 0, stream>>>(
        act, pwbf, pb, wgate, out, e);
  }
}
